// MultiHeadAttention_14998025797968
// MI455X (gfx1250) — matrix operand order repaired and checked by reading, not compiled, not run on hardware
//
#include <hip/hip_runtime.h>

// Problem constants (reference: B=2, S=2048, D=512, H=8, depth=64)
#define B_  2
#define S_  2048
#define D_  512
#define H_  8
#define DEPTH_ 64
#define M_  (B_ * S_)          // 4096 rows for the projection GEMMs
#define CHUNK_ 32              // flash s_k chunk
#define NCHUNK_ (S_ / CHUNK_)  // 64

typedef __attribute__((ext_vector_type(16))) __bf16        v16bf;
typedef __attribute__((ext_vector_type(8)))  float         v8f;
typedef __attribute__((ext_vector_type(4)))  unsigned int  u32x4;

struct Frag32B { u32x4 lo, hi; };   // 32 bytes == one v16bf fragment

__device__ __forceinline__ unsigned short f32_to_bf16(float f) {
  unsigned int u = __builtin_bit_cast(unsigned int, f);
  u += 0x7FFFu + ((u >> 16) & 1u);          // round-to-nearest-even
  return (unsigned short)(u >> 16);
}

// A-fragment (16x32 bf16): lane L holds row L%16, K = 16*(L>=16) .. +15 contiguous
__device__ __forceinline__ v16bf load_frag_a(const unsigned short* row, int hi) {
  Frag32B t;
  t.lo = *(const u32x4*)(row + 16 * hi);
  t.hi = *(const u32x4*)(row + 16 * hi + 8);
  return __builtin_bit_cast(v16bf, t);
}
// B-fragment (32x16 bf16): lane L holds one column, K = 16*(L>=16) .. +15 contiguous
__device__ __forceinline__ v16bf load_frag_b(const unsigned short* p) {
  Frag32B t;
  t.lo = *(const u32x4*)(p);
  t.hi = *(const u32x4*)(p + 8);
  return __builtin_bit_cast(v16bf, t);
}

__device__ __forceinline__ v8f wmma_bf16(v16bf a, v16bf b, v8f c) {
  return __builtin_amdgcn_wmma_f32_16x16x32_bf16(false, a, false, b, (short)0, c,
                                                 false, false);
}

__device__ __forceinline__ float rmax16(float v) {
  #pragma unroll
  for (int off = 1; off < 16; off <<= 1) v = fmaxf(v, __shfl_xor(v, off, 32));
  return v;
}
__device__ __forceinline__ float rsum16(float v) {
  #pragma unroll
  for (int off = 1; off < 16; off <<= 1) v += __shfl_xor(v, off, 32);
  return v;
}

// Async global->LDS copy (16B per lane). VDST = 32-bit LDS byte offset,
// VADDR = 64-bit global address, GV mode (saddr = off). Tracked by ASYNCcnt.
__device__ __forceinline__ void async_b128(const void* g, unsigned lds_off) {
  asm volatile("global_load_async_to_lds_b128 %0, %1, off"
               :: "v"(lds_off), "v"((unsigned long long)(uintptr_t)g)
               : "memory");
}

// ---------------- prep kernels ----------------
__global__ void cvt_bf16_kernel(const float* __restrict__ in,
                                unsigned short* __restrict__ out, int n) {
  int i = blockIdx.x * blockDim.x + threadIdx.x;
  if (i < n) out[i] = f32_to_bf16(in[i]);
}

// Wt[n][k] = bf16(W[k][n]), 512x512
__global__ void transpose_w_kernel(const float* __restrict__ w,
                                   unsigned short* __restrict__ wt) {
  int i = blockIdx.x * blockDim.x + threadIdx.x;
  if (i >= D_ * D_) return;
  int n = i / D_, k = i % D_;
  wt[n * D_ + k] = f32_to_bf16(w[k * D_ + n]);
}

// ---------------- projection / output GEMM ----------------
// C[16x16 tile] = A[M,512](bf16) @ Wt^T + bias, then *outScale. One wave/tile.
// mode 0: bf16 store into [B,H,S,64]       (Q with outScale=1/8, K)
// mode 1: bf16 store into [B,H,64,S]       (V transposed)
// mode 2: f32  store into [M,512]          (final output)
__global__ void __launch_bounds__(128)
gemm_bf16_kernel(const unsigned short* __restrict__ A,
                 const unsigned short* __restrict__ Wt,
                 const float* __restrict__ bias, void* __restrict__ dst,
                 int mode, float outScale) {
  const int wave = threadIdx.x >> 5;
  const int lane = threadIdx.x & 31;
  const int lo16 = lane & 15;
  const int hi   = lane >> 4;

  const int tiles_n = D_ / 16;                 // 32
  int tile = blockIdx.x * 4 + wave;            // 8192 tiles total
  int tm = tile / tiles_n;
  int tn = tile % tiles_n;

  const unsigned short* arow = A  + (tm * 16 + lo16) * D_;
  const unsigned short* wrow = Wt + (tn * 16 + lo16) * D_;

  v8f acc = {};
  #pragma unroll 4
  for (int k = 0; k < D_; k += 32) {
    v16bf af = load_frag_a(arow + k, hi);
    v16bf bf = load_frag_b(wrow + k + 16 * hi);
    acc = wmma_bf16(af, bf, acc);
  }

  const int gn = tn * 16 + lo16;               // output column
  const float bn = bias[gn];
  #pragma unroll
  for (int e = 0; e < 8; e++) {
    int gm = tm * 16 + e + 8 * hi;             // output row
    float v = (acc[e] + bn) * outScale;
    if (mode == 2) {
      ((float*)dst)[gm * D_ + gn] = v;
    } else {
      int b = gm >> 11, s = gm & (S_ - 1);     // S = 2048
      int h = gn >> 6,  d = gn & (DEPTH_ - 1);
      unsigned short bv = f32_to_bf16(v);
      if (mode == 0)
        ((unsigned short*)dst)[(((b * H_ + h) * S_ + s) * DEPTH_) + d] = bv;
      else
        ((unsigned short*)dst)[(((b * H_ + h) * DEPTH_ + d) * S_) + s] = bv;
    }
  }
}

// ---------------- flash attention ----------------
// Block = 4 waves = 64 query rows of one (b,h); wave owns 16 rows.
// K/V chunk staged in LDS by async DMA, double-buffered; all 4 waves share it.
__global__ void __launch_bounds__(128)
flash_attn_kernel(const unsigned short* __restrict__ Qb,
                  const unsigned short* __restrict__ Kb,
                  const unsigned short* __restrict__ Vt,
                  const float* __restrict__ mask,
                  unsigned short* __restrict__ Ctx) {
  __shared__ unsigned short Kt[2][CHUNK_ * DEPTH_];   // [32][64] bf16, 4KB each
  __shared__ unsigned short Vs[2][DEPTH_ * CHUNK_];   // [64][32] bf16, 4KB each
  __shared__ unsigned short Ptile[4][16 * 32];        // per-wave P staging

  const int wave = threadIdx.x >> 5;
  const int lane = threadIdx.x & 31;
  const int lo16 = lane & 15;
  const int hi   = lane >> 4;

  int tile = blockIdx.x * 4 + wave;            // 2048 q-tiles
  int qt = tile & (S_ / 16 - 1);               // 0..127
  int bh = tile >> 7;                          // 0..15 (same for all 4 waves)
  int b  = bh >> 3;
  int h  = bh & 7;
  int sq0 = qt * 16;

  const unsigned short* kcbase = Kb + (size_t)bh * S_ * DEPTH_;
  const unsigned short* vcbase = Vt + (size_t)bh * DEPTH_ * S_;

  // Per-wave share of one chunk's K (1KB contiguous) + V (16 rows x 64B).
  const int kElem = wave * 512 + lane * 16;           // element offset in 4KB K chunk
  const int vRow  = wave * 16 + (lane >> 1);          // depth row 0..63
  const int vHalf = (lane & 1) * 16;                  // element offset in 64B row

  auto issue_chunk = [&](int c, int bufIdx) {
    const unsigned short* gk = kcbase + (size_t)c * CHUNK_ * DEPTH_ + kElem;
    unsigned lk = (unsigned)(uintptr_t)&Kt[bufIdx][kElem];
    async_b128(gk,     lk);
    async_b128(gk + 8, lk + 16);
    const unsigned short* gv = vcbase + (size_t)vRow * S_ + c * CHUNK_ + vHalf;
    unsigned lv = (unsigned)(uintptr_t)&Vs[bufIdx][vRow * CHUNK_ + vHalf];
    async_b128(gv,     lv);
    async_b128(gv + 8, lv + 16);
  };

  // Q A-fragments for depth 0..31 and 32..63 (row = lo16); scale pre-folded.
  const unsigned short* qrow = Qb + ((bh * S_) + sq0 + lo16) * DEPTH_;
  v16bf qa0 = load_frag_a(qrow, hi);
  v16bf qa1 = load_frag_a(qrow + 32, hi);

  v8f o0 = {}, o1 = {}, o2 = {}, o3 = {};
  float m8[8], l8[8];
  #pragma unroll
  for (int e = 0; e < 8; e++) { m8[e] = -3.0e38f; l8[e] = 0.0f; }

  unsigned short* pw = Ptile[wave];
  const float* mbase = mask + (size_t)(b * S_ + sq0) * S_;

  issue_chunk(0, 0);

  for (int c = 0; c < NCHUNK_; ++c) {
    const int cur = c & 1;
    if (c + 1 < NCHUNK_) {
      issue_chunk(c + 1, cur ^ 1);
      asm volatile("s_wait_asynccnt 0x4" ::: "memory");  // chunk c retired
    } else {
      asm volatile("s_wait_asynccnt 0x0" ::: "memory");
    }
    __syncthreads();                                     // whole buffer valid

    const unsigned short* kt = Kt[cur];
    const unsigned short* vs = Vs[cur];

    // ---- energy: 16x32 tile = two C fragments; depth split 2x32 (from LDS)
    v8f e0 = {}, e1 = {};
    e0 = wmma_bf16(qa0, load_frag_b(kt + lo16 * DEPTH_ + 16 * hi), e0);
    e0 = wmma_bf16(qa1, load_frag_b(kt + lo16 * DEPTH_ + 32 + 16 * hi), e0);
    e1 = wmma_bf16(qa0, load_frag_b(kt + (16 + lo16) * DEPTH_ + 16 * hi), e1);
    e1 = wmma_bf16(qa1, load_frag_b(kt + (16 + lo16) * DEPTH_ + 32 + 16 * hi), e1);

    // ---- mask + online softmax (row = e + 8*hi, col = lo16 / lo16+16)
    const int sk0 = c * CHUNK_;
    #pragma unroll
    for (int e = 0; e < 8; e++) {
      const float* mrow = mbase + (size_t)(e + 8 * hi) * S_ + sk0;
      float mk0 = __builtin_nontemporal_load(mrow + lo16);
      float mk1 = __builtin_nontemporal_load(mrow + lo16 + 16);
      float v0 = fmaf(mk0, -1.0e10f, e0[e]);
      float v1 = fmaf(mk1, -1.0e10f, e1[e]);
      float cm = rmax16(fmaxf(v0, v1));
      float mn = fmaxf(m8[e], cm);
      float p0 = __expf(v0 - mn);
      float p1 = __expf(v1 - mn);
      float rs = rsum16(p0 + p1);
      float alpha = __expf(m8[e] - mn);
      l8[e] = l8[e] * alpha + rs;
      m8[e] = mn;
      o0[e] *= alpha; o1[e] *= alpha; o2[e] *= alpha; o3[e] *= alpha;
      pw[(e + 8 * hi) * 32 + lo16]      = f32_to_bf16(p0);
      pw[(e + 8 * hi) * 32 + lo16 + 16] = f32_to_bf16(p1);
    }
    asm volatile("s_wait_dscnt 0x0" ::: "memory");

    // ---- reload P as a 16x32 A-fragment (row = lo16)
    v16bf pa = load_frag_a(pw + lo16 * 32, hi);

    // ---- O += P @ V (V tile [64][32] in LDS; B-frag column = depth)
    o0 = wmma_bf16(pa, load_frag_b(vs + (lo16)      * CHUNK_ + 16 * hi), o0);
    o1 = wmma_bf16(pa, load_frag_b(vs + (16 + lo16) * CHUNK_ + 16 * hi), o1);
    o2 = wmma_bf16(pa, load_frag_b(vs + (32 + lo16) * CHUNK_ + 16 * hi), o2);
    o3 = wmma_bf16(pa, load_frag_b(vs + (48 + lo16) * CHUNK_ + 16 * hi), o3);

    __syncthreads();                                    // buffer reusable
  }

  // ---- epilogue: normalize and store ctx as bf16 [B,S,D] (merged heads)
  #pragma unroll
  for (int e = 0; e < 8; e++) {
    float inv = 1.0f / l8[e];
    int row = b * S_ + sq0 + e + 8 * hi;
    unsigned short* crow = Ctx + (size_t)row * D_ + h * DEPTH_;
    crow[lo16]      = f32_to_bf16(o0[e] * inv);
    crow[lo16 + 16] = f32_to_bf16(o1[e] * inv);
    crow[lo16 + 32] = f32_to_bf16(o2[e] * inv);
    crow[lo16 + 48] = f32_to_bf16(o3[e] * inv);
  }
}

// ---------------- host ----------------
extern "C" void kernel_launch(void* const* d_in, const int* in_sizes, int n_in,
                              void* d_out, int out_size, void* d_ws, size_t ws_size,
                              hipStream_t stream) {
  (void)in_sizes; (void)n_in; (void)out_size; (void)ws_size;
  const float* query = (const float*)d_in[0];
  const float* key   = (const float*)d_in[1];
  const float* value = (const float*)d_in[2];
  const float* mask  = (const float*)d_in[3];
  const float* wq = (const float*)d_in[4];  const float* bq = (const float*)d_in[5];
  const float* wk = (const float*)d_in[6];  const float* bk = (const float*)d_in[7];
  const float* wv = (const float*)d_in[8];  const float* bv = (const float*)d_in[9];
  const float* wo = (const float*)d_in[10]; const float* bo = (const float*)d_in[11];

  char* ws = (char*)d_ws;
  size_t off = 0;
  auto take = [&](size_t bytes) -> void* {
    void* p = ws + off;
    off += (bytes + 255) & ~(size_t)255;
    return p;
  };
  unsigned short* Xq  = (unsigned short*)take((size_t)M_ * D_ * 2);
  unsigned short* Xk  = (unsigned short*)take((size_t)M_ * D_ * 2);
  unsigned short* Xv  = (unsigned short*)take((size_t)M_ * D_ * 2);
  unsigned short* Wqt = (unsigned short*)take((size_t)D_ * D_ * 2);
  unsigned short* Wkt = (unsigned short*)take((size_t)D_ * D_ * 2);
  unsigned short* Wvt = (unsigned short*)take((size_t)D_ * D_ * 2);
  unsigned short* Wot = (unsigned short*)take((size_t)D_ * D_ * 2);
  unsigned short* Qb  = (unsigned short*)take((size_t)M_ * D_ * 2);
  unsigned short* Kb  = (unsigned short*)take((size_t)M_ * D_ * 2);
  unsigned short* Vtr = (unsigned short*)take((size_t)M_ * D_ * 2);
  unsigned short* Ctx = (unsigned short*)take((size_t)M_ * D_ * 2);

  const int nAct = M_ * D_;                 // 2M elements
  cvt_bf16_kernel<<<(nAct + 255) / 256, 256, 0, stream>>>(query, Xq, nAct);
  cvt_bf16_kernel<<<(nAct + 255) / 256, 256, 0, stream>>>(key,   Xk, nAct);
  cvt_bf16_kernel<<<(nAct + 255) / 256, 256, 0, stream>>>(value, Xv, nAct);
  transpose_w_kernel<<<(D_ * D_ + 255) / 256, 256, 0, stream>>>(wq, Wqt);
  transpose_w_kernel<<<(D_ * D_ + 255) / 256, 256, 0, stream>>>(wk, Wkt);
  transpose_w_kernel<<<(D_ * D_ + 255) / 256, 256, 0, stream>>>(wv, Wvt);
  transpose_w_kernel<<<(D_ * D_ + 255) / 256, 256, 0, stream>>>(wo, Wot);

  const int gemmBlocks = (M_ / 16) * (D_ / 16) / 4;     // 2048
  gemm_bf16_kernel<<<gemmBlocks, 128, 0, stream>>>(Xq, Wqt, bq, Qb, 0, 0.125f);
  gemm_bf16_kernel<<<gemmBlocks, 128, 0, stream>>>(Xk, Wkt, bk, Kb, 0, 1.0f);
  gemm_bf16_kernel<<<gemmBlocks, 128, 0, stream>>>(Xv, Wvt, bv, Vtr, 1, 1.0f);

  flash_attn_kernel<<<(B_ * H_ * (S_ / 16)) / 4, 128, 0, stream>>>(Qb, Kb, Vtr, mask, Ctx);

  gemm_bf16_kernel<<<gemmBlocks, 128, 0, stream>>>(Ctx, Wot, bo, d_out, 2, 1.0f);
}
